// GAT_82592221102847
// MI455X (gfx1250) — matrix operand order repaired and checked by reading, not compiled, not run on hardware
//
#include <hip/hip_runtime.h>
#include <math.h>

// ---------------------------------------------------------------------------
// GAT fused kernel set for MI455X (gfx1250, wave32, WMMA f16, async LDS).
// N=8192, F_IN=128, D=128.
// ---------------------------------------------------------------------------

#define GAT_N     8192
#define GAT_D     128
#define NEG_SLOPE 0.2f
#define MASK_FILL (-9.0e15f)

typedef __attribute__((ext_vector_type(16))) _Float16 v16h;
typedef __attribute__((ext_vector_type(2)))  __fp16   v2fp16;
typedef __attribute__((ext_vector_type(8)))  float    v8f;
typedef __attribute__((ext_vector_type(4)))  int      i4;
typedef __attribute__((ext_vector_type(4)))  float    f4;
typedef __attribute__((ext_vector_type(4)))  unsigned int u4;

// pointer types for the async global->LDS builtin (param types per hipcc
// diagnostic: vector-of-4-int pointers; AS1 source, AS3 destination)
typedef __attribute__((address_space(1))) i4* gbl_i4_p;
typedef __attribute__((address_space(3))) i4* lds_i4_p;

union Frag16 { unsigned int u[8]; v16h v; };

#if defined(__has_builtin)
#  if __has_builtin(__builtin_amdgcn_global_load_async_to_lds_b128)
#    define GAT_HAS_ASYNC_LDS 1
#  endif
#endif
#ifndef GAT_HAS_ASYNC_LDS
#  define GAT_HAS_ASYNC_LDS 0
#endif

// one v_cvt_pk_rtz_f16_f32, result as packed 32-bit
static __device__ __forceinline__ unsigned int pk16(float lo, float hi) {
  v2fp16 t = __builtin_amdgcn_cvt_pkrtz(lo, hi);
  return __builtin_bit_cast(unsigned int, t);
}

static __device__ __forceinline__ void wait_async_all() {
#if GAT_HAS_ASYNC_LDS
#  if __has_builtin(__builtin_amdgcn_s_wait_asynccnt)
  __builtin_amdgcn_s_wait_asynccnt(0);
#  else
  asm volatile("s_wait_asynccnt 0x0" ::: "memory");
#  endif
#endif
}

// ---------------------------------------------------------------------------
// Kernel 1: Wh = X @ Ws, stored TRANSPOSED as f16  (WhT[d][i], d<128, i<8192)
// One wave per 16-row block; 8 N-tiles x 4 K-steps of v_wmma_f32_16x16x32_f16.
// ---------------------------------------------------------------------------
__global__ __launch_bounds__(32)
void gat_wh_gemm(const float* __restrict__ X, const float* __restrict__ Ws,
                 _Float16* __restrict__ WhT)
{
  const int lane = threadIdx.x & 31;
  const int r    = lane & 15;   // A-fragment row / B-fragment column
  const int hi   = lane >> 4;   // lane-half selects K sub-range
  const int blk  = blockIdx.x;
  const int grow = blk * 16 + r;

  v8f zero = {0.f, 0.f, 0.f, 0.f, 0.f, 0.f, 0.f, 0.f};
  v8f acc[8];
#pragma unroll
  for (int t = 0; t < 8; ++t) acc[t] = zero;

  const float* xrow = X + (size_t)grow * GAT_D;

#pragma unroll
  for (int kk = 0; kk < 4; ++kk) {
    const int k0 = kk * 32;
    f4 x0 = *(const f4*)(xrow + k0 + hi * 8);
    f4 x1 = *(const f4*)(xrow + k0 + hi * 8 + 4);
    f4 x2 = *(const f4*)(xrow + k0 + 16 + hi * 8);
    f4 x3 = *(const f4*)(xrow + k0 + 16 + hi * 8 + 4);
    Frag16 af;
    af.u[0] = pk16(x0.x, x0.y); af.u[1] = pk16(x0.z, x0.w);
    af.u[2] = pk16(x1.x, x1.y); af.u[3] = pk16(x1.z, x1.w);
    af.u[4] = pk16(x2.x, x2.y); af.u[5] = pk16(x2.z, x2.w);
    af.u[6] = pk16(x3.x, x3.y); af.u[7] = pk16(x3.z, x3.w);

#pragma unroll
    for (int t = 0; t < 8; ++t) {
      Frag16 bf;
#pragma unroll
      for (int v = 0; v < 8; ++v) {
        const int k = k0 + hi * 8 + 2 * (v & 3) + 16 * (v >> 2);
        bf.u[v] = pk16(Ws[(size_t)k * GAT_D + 16 * t + r],
                       Ws[(size_t)(k + 1) * GAT_D + 16 * t + r]);
      }
      acc[t] = __builtin_amdgcn_wmma_f32_16x16x32_f16(
          false, af.v, false, bf.v, (short)0, acc[t], false, false);
    }
  }

#pragma unroll
  for (int t = 0; t < 8; ++t)
#pragma unroll
    for (int v = 0; v < 8; ++v) {
      const int col = 16 * t + r;
      const int row = blk * 16 + v + 8 * hi;
      WhT[(size_t)col * GAT_N + row] = (_Float16)acc[t][v];
    }
}

// ---------------------------------------------------------------------------
// Kernel 2: e1[i] = Wh[i,:].a[:128], e2[i] = Wh[i,:].a[128:]  (from WhT f16)
// ---------------------------------------------------------------------------
__global__ __launch_bounds__(256)
void gat_evec(const _Float16* __restrict__ WhT, const float* __restrict__ a,
              float* __restrict__ e1, float* __restrict__ e2)
{
  const int i = blockIdx.x * 256 + threadIdx.x;
  float s1 = 0.f, s2 = 0.f;
#pragma unroll 8
  for (int d = 0; d < GAT_D; ++d) {
    float v = (float)WhT[(size_t)d * GAT_N + i];  // coalesced across i
    s1 = fmaf(v, a[d], s1);
    s2 = fmaf(v, a[GAT_D + d], s2);
  }
  e1[i] = s1;
  e2[i] = s2;
}

// ---------------------------------------------------------------------------
// Kernel 3: fused flash-GAT. 4 waves / 64 rows per workgroup. The 32x128 Wh
// K-slice is staged into LDS (async global->LDS, double-buffered) and shared
// by all 4 waves; A mask streams non-temporally, register double-buffered.
// LDS slice layout: row d (0..127) at byte d*80, 32 f16 (64B) + 16B pad
// (stride 80B -> 16 distinct banks across lanes r=0..15).
// ---------------------------------------------------------------------------
#define SLICE_STRIDE 80
#define SLICE_BYTES  (128 * SLICE_STRIDE)

__global__ __launch_bounds__(128)
void gat_flash(const int* __restrict__ A, const _Float16* __restrict__ WhT,
               const float* __restrict__ e1, const float* __restrict__ e2,
               float* __restrict__ H)
{
  __shared__ __align__(16) unsigned char sWh[2][SLICE_BYTES];

  const int tid  = threadIdx.x;       // 0..127
  const int lane = tid & 31;
  const int r    = lane & 15;
  const int hi   = lane >> 4;
  const int wv   = tid >> 5;          // wave 0..3
  const int blk  = blockIdx.x;
  const int grow = blk * 64 + wv * 16 + r;
  const int kb   = hi * 8;            // K offset of this lane-half inside tile

  const float e1r = e1[grow];
  const int* Arow = A + (size_t)grow * GAT_N;

  // non-const base for the async builtin (it takes a non-const pointer)
  char* WhT_nc = (char*)const_cast<_Float16*>(WhT);

  // --- staging helper: copy WhT[0:128][j:j+32] into sWh[b] ------------------
  // 512 chunks of 16B; chunk c: d = c>>2, part = c&3.
  const int c0 = tid;  // chunks c0, c0+128, c0+256, c0+384
  auto stage = [&](int j, int b) {
#pragma unroll
    for (int q = 0; q < 4; ++q) {
      const int c    = c0 + 128 * q;
      const int d    = c >> 2;
      const int part = c & 3;
      char* src = WhT_nc + ((size_t)d * GAT_N + j + part * 8) * 2;
      unsigned char* dst = &sWh[b][d * SLICE_STRIDE + part * 16];
#if GAT_HAS_ASYNC_LDS
      __builtin_amdgcn_global_load_async_to_lds_b128(
          (gbl_i4_p)src, (lds_i4_p)dst, 0, 0);
#else
      u4 tmpv = *(const u4*)src;
      *(u4*)dst = tmpv;
#endif
    }
  };

  float m = -INFINITY;                // running row max
  float l = 0.0f;                     // running row sum
  v8f zero = {0.f, 0.f, 0.f, 0.f, 0.f, 0.f, 0.f, 0.f};
  v8f acc[8];
#pragma unroll
  for (int t = 0; t < 8; ++t) acc[t] = zero;

  // --- prologue: stage tile 0 into buffer 0; preload A/e2 tile 0 -----------
  stage(0, 0);
  i4 ca[4]; f4 cf[4];
  ca[0] = __builtin_nontemporal_load((const i4*)(Arow + kb));
  ca[1] = __builtin_nontemporal_load((const i4*)(Arow + kb + 4));
  ca[2] = __builtin_nontemporal_load((const i4*)(Arow + 16 + kb));
  ca[3] = __builtin_nontemporal_load((const i4*)(Arow + 16 + kb + 4));
  cf[0] = *(const f4*)(e2 + kb);
  cf[1] = *(const f4*)(e2 + kb + 4);
  cf[2] = *(const f4*)(e2 + 16 + kb);
  cf[3] = *(const f4*)(e2 + 16 + kb + 4);

  for (int j0 = 0; j0 < GAT_N; j0 += 32) {
    const int buf = (j0 >> 5) & 1;

    // my staging writes for `buf` done; all waves staged `buf` and finished
    // reading `buf^1` last iteration -> safe to consume `buf`, overwrite buf^1
    wait_async_all();
    __syncthreads();

    const int jn = (j0 + 32 < GAT_N) ? (j0 + 32) : 0;
    stage(jn, buf ^ 1);

    // --- next A/e2 tile (register double buffer; wrap keeps loads valid) ---
    i4 na[4]; f4 nf[4];
    na[0] = __builtin_nontemporal_load((const i4*)(Arow + jn + kb));
    na[1] = __builtin_nontemporal_load((const i4*)(Arow + jn + kb + 4));
    na[2] = __builtin_nontemporal_load((const i4*)(Arow + jn + 16 + kb));
    na[3] = __builtin_nontemporal_load((const i4*)(Arow + jn + 16 + kb + 4));
    nf[0] = *(const f4*)(e2 + jn + kb);
    nf[1] = *(const f4*)(e2 + jn + kb + 4);
    nf[2] = *(const f4*)(e2 + jn + 16 + kb);
    nf[3] = *(const f4*)(e2 + jn + 16 + kb + 4);
    if (j0 + 64 < GAT_N) __builtin_prefetch(Arow + j0 + 64 + kb, 0, 1);

    int   msk[16] = {ca[0].x, ca[0].y, ca[0].z, ca[0].w,
                     ca[1].x, ca[1].y, ca[1].z, ca[1].w,
                     ca[2].x, ca[2].y, ca[2].z, ca[2].w,
                     ca[3].x, ca[3].y, ca[3].z, ca[3].w};
    float ev[16]  = {cf[0].x, cf[0].y, cf[0].z, cf[0].w,
                     cf[1].x, cf[1].y, cf[1].z, cf[1].w,
                     cf[2].x, cf[2].y, cf[2].z, cf[2].w,
                     cf[3].x, cf[3].y, cf[3].z, cf[3].w};

    // --- masked leaky-relu scores -----------------------------------------
    float s[16];
#pragma unroll
    for (int i = 0; i < 16; ++i) {
      float e = e1r + ev[i];
      e = (e > 0.f) ? e : NEG_SLOPE * e;
      s[i] = (msk[i] > 0) ? e : MASK_FILL;
    }

    // --- online softmax stats (row split across lane pair L, L+16) --------
    float tmax = s[0];
#pragma unroll
    for (int i = 1; i < 16; ++i) tmax = fmaxf(tmax, s[i]);
    tmax = fmaxf(tmax, __shfl_xor(tmax, 16, 32));
    const float newm  = fmaxf(m, tmax);
    const float scale = __expf(m - newm);

    float p[16];
    float psum = 0.f;
#pragma unroll
    for (int i = 0; i < 16; ++i) {
      p[i] = __expf(s[i] - newm);
      psum += p[i];
    }
    psum += __shfl_xor(psum, 16, 32);
    l = l * scale + psum;
    m = newm;

    // --- P fragment (A-operand layout, f16) -------------------------------
    Frag16 pf;
#pragma unroll
    for (int v = 0; v < 4; ++v) {
      pf.u[v]     = pk16(p[2 * v],     p[2 * v + 1]);
      pf.u[4 + v] = pk16(p[8 + 2 * v], p[8 + 2 * v + 1]);
    }

    // --- rescale accumulators by per-row exp(m_old - m_new) ---------------
#pragma unroll
    for (int v = 0; v < 8; ++v) {
      const int src = ((wv << 5) + v + 8 * hi) << 2;  // within-wave lane v+8*hi
      const float sc =
          __int_as_float(__builtin_amdgcn_ds_bpermute(src, __float_as_int(scale)));
#pragma unroll
      for (int t = 0; t < 8; ++t) acc[t][v] *= sc;
    }

    // --- acc[t] += P (16x32) @ Wh[j0:j0+32, 16t:16t+16]  (f16 WMMA) -------
    // B fragment from LDS: lane column N = r -> slice row d = 16t+r,
    // 8 f16 along K at hi*8, then 8 f16 at 16+hi*8.
#pragma unroll
    for (int t = 0; t < 8; ++t) {
      const unsigned char* bsrc =
          &sWh[buf][(16 * t + r) * SLICE_STRIDE + hi * 16];
      u4 b0 = *(const u4*)bsrc;
      u4 b1 = *(const u4*)(bsrc + 32);
      Frag16 bfr;
      bfr.u[0] = b0.x; bfr.u[1] = b0.y; bfr.u[2] = b0.z; bfr.u[3] = b0.w;
      bfr.u[4] = b1.x; bfr.u[5] = b1.y; bfr.u[6] = b1.z; bfr.u[7] = b1.w;
      acc[t] = __builtin_amdgcn_wmma_f32_16x16x32_f16(
          false, pf.v, false, bfr.v, (short)0, acc[t], false, false);
    }

    // --- rotate A/e2 double buffer ----------------------------------------
#pragma unroll
    for (int q = 0; q < 4; ++q) { ca[q] = na[q]; cf[q] = nf[q]; }
  }

  // --- finalize: H[row] = acc[row] / l[row] --------------------------------
  const float inv = 1.0f / l;
#pragma unroll
  for (int v = 0; v < 8; ++v) {
    const int src = ((wv << 5) + v + 8 * hi) << 2;
    const float iv =
        __int_as_float(__builtin_amdgcn_ds_bpermute(src, __float_as_int(inv)));
    const int row = blk * 64 + wv * 16 + v + 8 * hi;
#pragma unroll
    for (int t = 0; t < 8; ++t)
      __builtin_nontemporal_store(acc[t][v] * iv,
                                  H + (size_t)row * GAT_D + 16 * t + r);
  }
}

// ---------------------------------------------------------------------------
// Launch
// ---------------------------------------------------------------------------
extern "C" void kernel_launch(void* const* d_in, const int* in_sizes, int n_in,
                              void* d_out, int out_size, void* d_ws, size_t ws_size,
                              hipStream_t stream) {
  (void)in_sizes; (void)n_in; (void)out_size; (void)ws_size;

  const int*   A  = (const int*)d_in[0];     // [8192, 8192] int32
  const float* X  = (const float*)d_in[1];   // [8192, 128]  f32
  const float* Ws = (const float*)d_in[2];   // [128, 128]   f32
  const float* a  = (const float*)d_in[3];   // [256, 1]     f32
  float*       H  = (float*)d_out;           // [8192, 128]  f32

  // scratch: WhT f16 (2 MB) | e1 (32 KB) | e2 (32 KB)
  _Float16* WhT = (_Float16*)d_ws;
  float* e1 = (float*)((char*)d_ws + (size_t)GAT_D * GAT_N * sizeof(_Float16));
  float* e2 = e1 + GAT_N;

  gat_wh_gemm<<<GAT_N / 16, 32, 0, stream>>>(X, Ws, WhT);
  gat_evec<<<GAT_N / 256, 256, 0, stream>>>(WhT, a, e1, e2);
  gat_flash<<<GAT_N / 64, 128, 0, stream>>>(A, WhT, e1, e2, H);
}
